// WorkingMemory_53979148976723
// MI455X (gfx1250) — matrix operand order repaired and checked by reading, not compiled, not run on hardware
//
#include <hip/hip_runtime.h>

// ---------------- problem constants ----------------
#define TT   512
#define BB   128
#define HH   300
#define MMC  20          // NUM_CELLS
#define MLP  300
#define EPSF 1e-8f
#define DECAYF 0.98f

// padded geometry
#define MEMC 304         // padded H (row stride of fp32 state arrays)
#define AKP  928         // padded K for pair GEMM: 29 k-tiles * 32
#define MKP  320         // padded K for mem GEMM : 10 k-tiles * 32
#define NPAD 304         // padded N (19 n-tiles * 16)

typedef __attribute__((ext_vector_type(16))) __bf16 v16bf;
typedef __attribute__((ext_vector_type(8)))  __bf16 v8bf;
typedef __attribute__((ext_vector_type(8)))  float  v8f;

// ---------------- WMMA helper ----------------
__device__ __forceinline__ v8f wmma_bf16(v16bf a, v16bf b, v8f c) {
  // D = A(16x32 bf16) * B(32x16 bf16) + C(16x16 f32)
  return __builtin_amdgcn_wmma_f32_16x16x32_bf16(false, a, false, b, (short)0, c,
                                                 false, false);
}

// B fragment: bf16 matrix stored N-major [n][K], kstride halves per row.
// Layout: half i <-> K = kt*32 + 8*(lane>=16) + (i<8 ? i : i+8), col n = lane&15.
__device__ __forceinline__ v16bf load_bfrag(const __bf16* __restrict__ Bt,
                                            int n, int kt, int kstride) {
  int lane = threadIdx.x & 31;
  const __bf16* p = Bt + (size_t)n * kstride + kt * 32 + 8 * (lane >> 4);
  v8bf lo = *(const v8bf*)p;
  v8bf hi = *(const v8bf*)(p + 16);
  v16bf r;
#pragma unroll
  for (int i = 0; i < 8; i++) { r[i] = lo[i]; r[i + 8] = hi[i]; }
  return r;
}

// A fragment from LDS: bf16 matrix [m][K], kstride halves per row.
// Layout: half i <-> K = kt*32 + 8*(lane>=16) + (i<8 ? i : i+8), row m = given.
__device__ __forceinline__ v16bf load_afrag_lds(const __bf16* A, int m, int kt,
                                                int kstride) {
  int lane = threadIdx.x & 31;
  const __bf16* p = A + m * kstride + kt * 32 + 8 * (lane >> 4);
  v8bf r0 = *(const v8bf*)p;         // K offsets +0..7
  v8bf r1 = *(const v8bf*)(p + 16);  // K offsets +16..23
  v16bf r;
#pragma unroll
  for (int i = 0; i < 8; i++) { r[i] = r0[i]; r[i + 8] = r1[i]; }
  return r;
}

// A fragment from a global fp32 row (converted to bf16 on the fly, K<HH guard).
__device__ __forceinline__ v16bf load_afrag_f32(const float* __restrict__ rowptr,
                                                int kt) {
  int lane = threadIdx.x & 31;
  int kb = kt * 32 + 8 * (lane >> 4);
  v16bf r;
#pragma unroll
  for (int i = 0; i < 16; i++) {
    int k = kb + (i < 8 ? i : i + 8);
    float f = (k < HH) ? rowptr[k] : 0.0f;
    r[i] = (__bf16)f;
  }
  return r;
}

// wave32 reductions
__device__ __forceinline__ float wave_sum(float v) {
#pragma unroll
  for (int d = 16; d; d >>= 1) v += __shfl_xor(v, d, 32);
  return v;
}
__device__ __forceinline__ float wave_max(float v) {
#pragma unroll
  for (int d = 16; d; d >>= 1) v = fmaxf(v, __shfl_xor(v, d, 32));
  return v;
}
__device__ __forceinline__ int wave_min_i(int v) {
#pragma unroll
  for (int d = 16; d; d >>= 1) { int o = __shfl_xor(v, d, 32); v = o < v ? o : v; }
  return v;
}
// sum over a 16-lane half (columns of one C tile)
__device__ __forceinline__ float half_sum(float v) {
#pragma unroll
  for (int d = 8; d; d >>= 1) v += __shfl_xor(v, d, 32);
  return v;
}

// ---------------- kernel 0: weight transpose/convert to bf16 ----------------
// We1t [NPAD][MKP] : We1t[n][k] = We1[k][n]
// Ws1t [NPAD][AKP] : Ws1t[n][k] = Ws1[k][n]      (k<900; row 900 folded into init)
// Wut  [NPAD][MKP] : Wut [n][k] = Wu[300+k][n]   (bottom half of Wu)
__global__ void wm_convert_kernel(const float* __restrict__ We1,
                                  const float* __restrict__ Ws1,
                                  const float* __restrict__ Wu,
                                  __bf16* __restrict__ We1t,
                                  __bf16* __restrict__ Ws1t,
                                  __bf16* __restrict__ Wut) {
  int idx = blockIdx.x * blockDim.x + threadIdx.x;
  if (idx >= 3 * NPAD) return;
  int which = idx / NPAD, n = idx - which * NPAD;
  if (which == 0) {
    for (int k = 0; k < MKP; k++) {
      float v = (n < MLP && k < HH) ? We1[k * MLP + n] : 0.0f;
      We1t[(size_t)n * MKP + k] = (__bf16)v;
    }
  } else if (which == 1) {
    for (int k = 0; k < AKP; k++) {
      float v = (n < MLP && k < 900) ? Ws1[k * MLP + n] : 0.0f;
      Ws1t[(size_t)n * AKP + k] = (__bf16)v;
    }
  } else {
    for (int k = 0; k < MKP; k++) {
      float v = (n < HH && k < HH) ? Wu[(size_t)(HH + k) * HH + n] : 0.0f;
      Wut[(size_t)n * MKP + k] = (__bf16)v;
    }
  }
}

// ---------------- kernel 1: hoisted entity-score GEMM (all T*B rows) ----------
// z = relu(h @ We1 + be1); s = z @ We2 + be2; ent = sigmoid(s)*mask
// One wave per 16-row tile; 4096 waves total. B fragments are double-buffered:
// the (kt+1) load issues before the WMMA consuming kt, so the wait before each
// WMMA can leave one load in flight instead of draining to zero.
__global__ __launch_bounds__(256) void wm_ent_kernel(
    const float* __restrict__ hidden, const float* __restrict__ mask,
    const __bf16* __restrict__ We1t, const float* __restrict__ be1,
    const float* __restrict__ We2, const float* __restrict__ be2,
    float* __restrict__ ent_prob, float* __restrict__ out_ent) {
  int gwave = (blockIdx.x * blockDim.x + threadIdx.x) >> 5;
  int lane = threadIdx.x & 31;
  int row0 = gwave * 16;
  if (row0 >= TT * BB) return;
  int m16 = lane & 15, hi = lane >> 4;
  const float* rowptr = hidden + (size_t)(row0 + m16) * HH;

  v16bf afrag[10];
#pragma unroll
  for (int kt = 0; kt < 10; kt++) afrag[kt] = load_afrag_f32(rowptr, kt);

  float zacc[8];
#pragma unroll
  for (int v = 0; v < 8; v++) zacc[v] = 0.0f;

  for (int nt = 0; nt < 19; nt++) {
    int n = nt * 16 + m16;
    v8f acc = {};
    v16bf bb = load_bfrag(We1t, n, 0, MKP);
#pragma unroll
    for (int kt = 0; kt < 10; kt++) {
      v16bf bn = load_bfrag(We1t, n, (kt < 9) ? kt + 1 : 9, MKP);
      acc = wmma_bf16(afrag[kt], bb, acc);
      bb = bn;
    }
    if (n < MLP) {
      float b1 = be1[n], w2 = We2[n];
#pragma unroll
      for (int v = 0; v < 8; v++) {
        float z = acc[v] + b1;
        z = z > 0.0f ? z : 0.0f;
        zacc[v] += z * w2;  // lane holds (row = v+8*hi, col = n)
      }
    }
  }
#pragma unroll
  for (int v = 0; v < 8; v++) zacc[v] = half_sum(zacc[v]);
  if (m16 == 0) {  // lanes 0 (rows 0..7) and 16 (rows 8..15)
#pragma unroll
    for (int v = 0; v < 8; v++) {
      int row = row0 + v + 8 * hi;
      float s = zacc[v] + be2[0];
      float p = (1.0f / (1.0f + expf(-s))) * mask[row];
      ent_prob[row] = p;
      out_ent[row] = p * (1.0f - EPSF) + EPSF;
    }
  }
}

// ---------------- kernel 2: recurrent scan, one block per batch element ------
__global__ __launch_bounds__(256) void wm_scan_kernel(
    const float* __restrict__ hidden, const float* __restrict__ ent_prob,
    const __bf16* __restrict__ Ws1t, const float* __restrict__ bs1,
    const float* __restrict__ Ws2, const float* __restrict__ bs2,
    const __bf16* __restrict__ Wut, const float* __restrict__ Wu,
    const float* __restrict__ bu, const float* __restrict__ Ws1,
    float* __restrict__ out_usage, float* __restrict__ out_coref,
    float* __restrict__ out_ow) {
  __shared__ float memS[MMC * MEMC];          // fp32 recurrent memory
  __shared__ float candS[MMC * MEMC];         // tanh candidate
  __shared__ float hS[MEMC];                  // current h
  __shared__ float huS[MEMC];                 // h @ Wu_top
  __shared__ __align__(16) __bf16 Abf[32 * AKP];  // pair matrix, bf16, padded
  __shared__ __align__(16) __bf16 Mbf[32 * MKP];  // mem matrix,  bf16, padded
  __shared__ float simP[19 * 32];             // per-ntile sim partials
  __shared__ float usageS[32];
  __shared__ float owS[32], indvS[32];

  const int b = blockIdx.x;
  const int tid = threadIdx.x;
  const int lane = tid & 31, wid = tid >> 5;
  const int m16 = lane & 15, hi = lane >> 4;

  for (int i = tid; i < MMC * MEMC; i += 256) memS[i] = 0.0f;
  if (tid < 32) usageS[tid] = 0.0f;
  __syncthreads();

  for (int t = 0; t < TT; t++) {
    // ---- stage h; prefetch next step's row across the serial dependency ----
    const float* hp = hidden + ((size_t)t * BB + b) * HH;
    for (int i = tid; i < MEMC; i += 256) hS[i] = (i < HH) ? hp[i] : 0.0f;
    if (t + 1 < TT) {
      const float* hn = hidden + ((size_t)(t + 1) * BB + b) * HH;
      if (tid < 10) __builtin_prefetch(hn + tid * 32, 0, 3);  // global_prefetch_b8
    }
    __syncthreads();

    // ---- build bf16 operands: Mbf = mem, Abf = [mem | h | h*mem] ----
    for (int i = tid; i < 32 * MKP; i += 256) {
      int m = i / MKP, k = i - m * MKP;
      float v = (m < MMC && k < HH) ? memS[m * MEMC + k] : 0.0f;
      Mbf[i] = (__bf16)v;
    }
    for (int i = tid; i < 32 * AKP; i += 256) {
      int m = i / AKP, k = i - m * AKP;
      float v = 0.0f;
      if (m < MMC) {
        if (k < 300)      v = memS[m * MEMC + k];
        else if (k < 600) v = hS[k - 300];
        else if (k < 900) v = hS[k - 600] * memS[m * MEMC + (k - 600)];
      }
      Abf[i] = (__bf16)v;
    }
    // ---- hu = h @ Wu_top (rank-1 part of cand, shared by all cells) ----
    for (int n = tid; n < HH; n += 256) {
      float s = 0.0f;
      for (int k = 0; k < HH; k++) s += hS[k] * Wu[(size_t)k * HH + n];
      huS[n] = s;
    }
    __syncthreads();

    // ---- sim GEMM: relu(pair @ Ws1 + bs1) @ Ws2 ; usage column folded.
    //      One job per N-tile covers BOTH M-tiles (one B load, two WMMAs). ----
    for (int nt = wid; nt < 19; nt += 8) {
      int n = nt * 16 + m16;
      float b1 = 0.0f, w900 = 0.0f, w2 = 0.0f;
      if (n < MLP) { b1 = bs1[n]; w900 = Ws1[(size_t)900 * MLP + n]; w2 = Ws2[n]; }
      v8f acc0, acc1;
#pragma unroll
      for (int v = 0; v < 8; v++) {
        acc0[v] = b1 + usageS[v + 8 * hi] * w900;
        acc1[v] = b1 + usageS[16 + v + 8 * hi] * w900;
      }
      v16bf bb = load_bfrag(Ws1t, n, 0, AKP);
      for (int kt = 0; kt < 29; kt++) {
        v16bf bn = load_bfrag(Ws1t, n, (kt < 28) ? kt + 1 : 28, AKP);
        v16bf a0 = load_afrag_lds(Abf, m16, kt, AKP);
        v16bf a1 = load_afrag_lds(Abf, 16 + m16, kt, AKP);
        acc0 = wmma_bf16(a0, bb, acc0);
        acc1 = wmma_bf16(a1, bb, acc1);
        bb = bn;
      }
#pragma unroll
      for (int v = 0; v < 8; v++) {
        float z0 = acc0[v] > 0.0f ? acc0[v] : 0.0f;
        float z1 = acc1[v] > 0.0f ? acc1[v] : 0.0f;
        float p0 = (n < MLP) ? z0 * w2 : 0.0f;
        float p1 = (n < MLP) ? z1 * w2 : 0.0f;
        p0 = half_sum(p0);
        p1 = half_sum(p1);
        if (m16 == 0) {
          simP[nt * 32 + (v + 8 * hi)] = p0;
          simP[nt * 32 + (16 + v + 8 * hi)] = p1;
        }
      }
    }

    // ---- cand GEMM: tanh(mem @ Wu_bot + hu + bu), dual M-tile per job ----
    for (int nt = wid; nt < 19; nt += 8) {
      int n = nt * 16 + m16;
      v8f acc0 = {}, acc1 = {};
      v16bf bb = load_bfrag(Wut, n, 0, MKP);
      for (int kt = 0; kt < 10; kt++) {
        v16bf bn = load_bfrag(Wut, n, (kt < 9) ? kt + 1 : 9, MKP);
        v16bf a0 = load_afrag_lds(Mbf, m16, kt, MKP);
        v16bf a1 = load_afrag_lds(Mbf, 16 + m16, kt, MKP);
        acc0 = wmma_bf16(a0, bb, acc0);
        acc1 = wmma_bf16(a1, bb, acc1);
        bb = bn;
      }
      if (n < HH) {
        float add = huS[n] + bu[n];
#pragma unroll
        for (int v = 0; v < 8; v++) {
          int m0 = v + 8 * hi;            // 0..15, always a valid cell row
          candS[m0 * MEMC + n] = tanhf(acc0[v] + add);
          int m1 = 16 + v + 8 * hi;       // 16..31, valid only below MMC
          if (m1 < MMC) candS[m1 * MEMC + n] = tanhf(acc1[v] + add);
        }
      }
    }
    __syncthreads();

    // ---- softmax / routing / usage update: one wave, lane = cell ----
    if (wid == 0) {
      int m = lane;
      bool valid = m < MMC;
      float simv = -1e30f, us = 0.0f;
      if (valid) {
        float s = bs2[0];
#pragma unroll
        for (int nt = 0; nt < 19; nt++) s += simP[nt * 32 + m];
        simv = s;
        us = usageS[m];
      }
      float cmask = (us > 0.0f) ? 1.0f : 0.0f;
      float comb = valid ? (cmask * simv + (1.0f - cmask) * (-10000.0f)) : -1e30f;
      // softmax over 21 slots (20 cells + null slot with logit 0, mult 1)
      float mx = fmaxf(wave_max(comb), 0.0f);
      float e = valid ? expf(comb - mx) : 0.0f;
      float eM = expf(0.0f - mx);
      float se = wave_sum(e) + eM;
      float prob = e / se, probM = eM / se;
      float masked = prob * cmask;
      float smask = wave_sum(masked) + probM;
      float ep = ent_prob[t * BB + b];
      float indv = ep * (masked / (smask + EPSF));
      float ow_base = ep * (probM / (smask + EPSF));
      // nsim over the 20 cells
      float mxs = wave_max(simv);
      float es = valid ? expf(simv - mxs) : 0.0f;
      float nsim = es / wave_sum(es);
      float ow_score = valid ? (((us == 0.0f) ? nsim * 100000.0f : 0.0f) +
                                (1.0f - us))
                             : -1e30f;
      float mv = wave_max(ow_score);
      // noise only tie-breaks exact maxima (measure-zero); take first index
      int cidx = (valid && ow_score == mv) ? m : 1000;
      cidx = wave_min_i(cidx);
      float ow = (m == cidx) ? ow_base : 0.0f;
      float nus = valid ? fminf(1.0f, ow + indv + DECAYF * us) : 0.0f;
      if (valid) {
        size_t o = ((size_t)t * BB + b) * MMC + m;
        out_usage[o] = nus;
        out_coref[o] = indv * (1.0f - EPSF) + EPSF;
        out_ow[o]    = ow   * (1.0f - EPSF) + EPSF;
        usageS[m] = nus;
        owS[m] = ow;
        indvS[m] = indv;
      }
    }
    __syncthreads();

    // ---- memory update ----
    for (int i = tid; i < MMC * HH; i += 256) {
      int m = i / HH, c = i - m * HH;
      float ow = owS[m], iv = indvS[m];
      int idx = m * MEMC + c;
      memS[idx] = ow * hS[c] + (1.0f - ow - iv) * memS[idx] + iv * candS[idx];
    }
    __syncthreads();
  }
}

// ---------------- launch ----------------
extern "C" void kernel_launch(void* const* d_in, const int* in_sizes, int n_in,
                              void* d_out, int out_size, void* d_ws, size_t ws_size,
                              hipStream_t stream) {
  const float* hidden = (const float*)d_in[0];
  const float* imask  = (const float*)d_in[1];
  const float* We1    = (const float*)d_in[2];
  const float* be1    = (const float*)d_in[3];
  const float* We2    = (const float*)d_in[4];
  const float* be2    = (const float*)d_in[5];
  const float* Ws1    = (const float*)d_in[6];
  const float* bs1    = (const float*)d_in[7];
  const float* Ws2    = (const float*)d_in[8];
  const float* bs2    = (const float*)d_in[9];
  const float* Wu     = (const float*)d_in[10];
  const float* bu     = (const float*)d_in[11];

  char* ws = (char*)d_ws;
  __bf16* We1t = (__bf16*)(ws + 0);        // 304*320*2 = 194560 B
  __bf16* Ws1t = (__bf16*)(ws + 194560);   // 304*928*2 = 564224 B
  __bf16* Wut  = (__bf16*)(ws + 758784);   // 304*320*2 = 194560 B
  float*  entp = (float*)(ws + 953344);    // 65536*4   = 262144 B

  float* out       = (float*)d_out;
  float* out_ent   = out;                          // (T,B,1)
  float* out_usage = out + TT * BB;                // (T,B,M)
  float* out_coref = out_usage + TT * BB * MMC;    // (T,B,M)
  float* out_ow    = out_coref + TT * BB * MMC;    // (T,B,M)

  wm_convert_kernel<<<(3 * NPAD + 255) / 256, 256, 0, stream>>>(
      We1, Ws1, Wu, We1t, Ws1t, Wut);
  wm_ent_kernel<<<(TT * BB / 16 + 7) / 8, 256, 0, stream>>>(
      hidden, imask, We1t, be1, We2, be2, entp, out_ent);
  wm_scan_kernel<<<BB, 256, 0, stream>>>(
      hidden, entp, Ws1t, bs1, Ws2, bs2, Wut, Wu, bu, Ws1,
      out_usage, out_coref, out_ow);
}
